// TextureWarpingModule_21577915695504
// MI455X (gfx1250) — matrix operand order repaired and checked by reading, not compiled, not run on hardware
//
#include <hip/hip_runtime.h>
#include <hip/hip_bf16.h>
#include <math.h>

// ---------------- problem constants ----------------
#define HWD   4096            // 64*64
#define NTOT  16384           // B*H*W = 4*4096
#define LDA   40              // LDS row stride (ushorts) for A tile: 80B, 16B-aligned
#define LDB   40              // LDS row stride (ushorts) for B tile

typedef __attribute__((ext_vector_type(16))) __bf16          v16bf;
typedef __attribute__((ext_vector_type(8)))  float           v8f;
typedef __attribute__((ext_vector_type(16))) unsigned short  v16u;
typedef __attribute__((ext_vector_type(8)))  unsigned short  v8u;

// fp32 -> bf16 round-to-nearest-even (pure integer; no __bf16 arithmetic)
static __device__ __forceinline__ unsigned short f2bf(float f) {
  unsigned int u = __float_as_uint(f);
  unsigned int r = u + 0x7FFFu + ((u >> 16) & 1u);
  return (unsigned short)(r >> 16);
}

// =====================================================================
// Per-tap implicit GEMM:  Out[m][n] (+)= sum_k W[m][k] * X[k][shift(n)]
//   n = b*4096 + h*64 + w ; X read at (h+dh, w+dw), zero-padded.
//   W element (m,kc) at Wgt[(m*Cin+kc)*KSTRIDE + tap]
//   SPLIT: kc < C0 -> X0 else X1 (concat input; uniform per K-step, C0%32==0).
//   !SPLIT: single input X0, B pointer is a pure strided induction variable.
//   accIn == nullptr : C = bias (or 0); else C loaded from accIn.
//
// Block: 256 threads / 8 waves; tile 64(M) x 128(N).
// Wave (wm 0..3, wn 0..1): 16(M) x 64(N) = one A fragment, 4 B fragments,
// 4 x v_wmma_f32_16x16x32_bf16 per K-step.
//
// No predicated loads: out-of-range lanes load a clamped in-bounds address
// (finite value) and multiply by a 0/1 mask -> no exec-mask churn.
// =====================================================================
template <int KSTRIDE, bool SPLIT>
__global__ __launch_bounds__(256)
void gemm_tap_kernel(const float* __restrict__ Wgt, int tap,
                     int M, int Cin, int C0,
                     const float* __restrict__ X0, const float* __restrict__ X1,
                     int dh, int dw,
                     const float* __restrict__ bias,
                     const float* __restrict__ accIn,
                     float* __restrict__ Out)
{
  __shared__ unsigned short Asm[64 * LDA];    // [m_local][k] bf16 bits
  __shared__ unsigned short Bsm[128 * LDB];   // [n_local][k] bf16 bits

  const int tid  = threadIdx.x;
  const int lane = tid & 31;
  const int wave = tid >> 5;
  const int wm   = wave >> 1;        // 0..3
  const int wn   = wave & 1;         // 0..1
  const int l16  = lane & 15;
  const int hi   = lane >> 4;

  const int mBase = blockIdx.y * 64;
  const int nBase = blockIdx.x * 128;     // grid.x = NTOT/128
  const int bIdx  = nBase >> 12;          // batch index  (block-uniform)
  const int hw0   = nBase & 4095;         // multiple of 128
  const int h0    = hw0 >> 6;             // first of the two rows this block covers
  const bool fullM = (mBase + 64) <= M;   // uniform: no partial-M guards needed

  // ---------- per-thread loader precompute (loop-invariant) ----------
  // A loader: one LDS row-quarter: row a_row, k in [a_kq, a_kq+8)
  const int a_row = tid >> 2;
  const int a_kq  = (tid & 3) * 8;
  const float a_okf = ((mBase + a_row) < M) ? 1.f : 0.f;
  const int a_mrow  = ((mBase + a_row) < M) ? (mBase + a_row) : 0;   // clamped, in-bounds
  const float* ap = Wgt + a_mrow * Cin * KSTRIDE + a_kq * KSTRIDE + tap;

  // B loader: one pixel (n_local), 16 channels [b_kq, b_kq+16)
  const int b_nl  = tid >> 1;             // 0..127
  const int b_kq  = (tid & 1) * 16;       // 0 or 16
  const int b_hh  = h0 + (b_nl >> 6) + dh;
  const int b_ww  = (b_nl & 63) + dw;
  const bool b_ok = ((unsigned)b_hh < 64u) && ((unsigned)b_ww < 64u);
  const float b_okf = b_ok ? 1.f : 0.f;
  const int b_pix = b_ok ? (b_hh * 64 + b_ww) : 0;                    // clamped, in-bounds
  const float* x0p = X0 + (bIdx * C0 + b_kq) * HWD + b_pix;           // + k0*HWD
  const float* x1p = SPLIT ? (X1 + (bIdx * (Cin - C0) + b_kq) * HWD + b_pix) : nullptr;

  // ---------- accumulators: 4 tiles of 16x16 ----------
  v8f acc[4];
  {
    const int mrow0 = mBase + wm * 16;                 // + i + 8*hi
    const int obase = (bIdx * M + mrow0) * HWD + hw0;  // + (i+8*hi)*HWD + nl
#pragma unroll
    for (int t = 0; t < 4; ++t) {
      const int nl = wn * 64 + t * 16 + l16;
#pragma unroll
      for (int i = 0; i < 8; ++i) {
        const int mo = i + 8 * hi;                     // C/D layout: VGPR i -> M=i (+8 hi half)
        float a0 = 0.f;
        if (fullM || (mrow0 + mo) < M) {
          if (accIn)     a0 = accIn[obase + mo * HWD + nl];
          else if (bias) a0 = bias[mrow0 + mo];
        }
        acc[t][i] = a0;
      }
    }
  }

  // B pointer as induction variable for the common (!SPLIT) case
  const float* xp_iv = x0p;

  // ---------- K loop ----------
  for (int k0 = 0; k0 < Cin; k0 += 32) {
    __syncthreads();

    // stage A: 8 unconditional weight loads -> one 16B ds_store_b128
    {
      v8u pk;
#pragma unroll
      for (int j = 0; j < 8; ++j)
        pk[j] = f2bf(ap[j * KSTRIDE] * a_okf);
      *(v8u*)&Asm[a_row * LDA + a_kq] = pk;
      ap += 32 * KSTRIDE;
    }
    // stage B: 16 unconditional channel-strided loads -> two 16B ds_store_b128
    {
      const float* xp;
      if (SPLIT) {
        xp = (k0 < C0) ? (x0p + k0 * HWD) : (x1p + (k0 - C0) * HWD);
        const int lim = (k0 < C0) ? C0 : Cin;
        if (k0 + 32 < lim) __builtin_prefetch(xp + 32 * HWD, 0, 3);
      } else {
        xp = xp_iv;
        if (k0 + 32 < Cin) __builtin_prefetch(xp + 32 * HWD, 0, 3);
      }
      v8u lo, hi8;
#pragma unroll
      for (int j = 0; j < 8; ++j) {
        lo[j]  = f2bf(xp[j * HWD] * b_okf);
        hi8[j] = f2bf(xp[(j + 8) * HWD] * b_okf);
      }
      *(v8u*)&Bsm[b_nl * LDB + b_kq]     = lo;
      *(v8u*)&Bsm[b_nl * LDB + b_kq + 8] = hi8;
      if (!SPLIT) xp_iv += 32 * HWD;
    }
    __syncthreads();

    // A fragment (16x32): lane<16 -> K 0-7|16-23 of row l16; lane>=16 -> K 8-15|24-31
    const v8u* Ar = (const v8u*)&Asm[(wm * 16 + l16) * LDA];
    v8u a_lo = Ar[hi];
    v8u a_hi = Ar[hi + 2];
    v16u au;
#pragma unroll
    for (int i = 0; i < 8; ++i) { au[i] = a_lo[i]; au[8 + i] = a_hi[i]; }
    const v16bf a = __builtin_bit_cast(v16bf, au);

    // 4 B fragments (32x16 each), one WMMA per fragment; A reused 4x
#pragma unroll
    for (int t = 0; t < 4; ++t) {
      const v8u* Br = (const v8u*)&Bsm[(wn * 64 + t * 16 + l16) * LDB];
      v8u q0 = Br[hi], q1 = Br[hi + 2];
      v16u bu;
#pragma unroll
      for (int i = 0; i < 8; ++i) { bu[i] = q0[i]; bu[8 + i] = q1[i]; }
      const v16bf bf = __builtin_bit_cast(v16bf, bu);
      acc[t] = __builtin_amdgcn_wmma_f32_16x16x32_bf16(false, a, false, bf, (short)0,
                                                       acc[t], false, false);
    }
  }

  // ---------- store (lanes 0..15 -> 64B contiguous) ----------
  {
    const int mrow0 = mBase + wm * 16;
    float* op = Out + (bIdx * M + mrow0) * HWD + hw0;
    if (fullM) {
#pragma unroll
      for (int t = 0; t < 4; ++t) {
        const int nl = wn * 64 + t * 16 + l16;
#pragma unroll
        for (int i = 0; i < 8; ++i)
          op[(i + 8 * hi) * HWD + nl] = acc[t][i];
      }
    } else {
#pragma unroll
      for (int t = 0; t < 4; ++t) {
        const int nl = wn * 64 + t * 16 + l16;
#pragma unroll
        for (int i = 0; i < 8; ++i) {
          const int mo = i + 8 * hi;
          if ((mrow0 + mo) < M) op[mo * HWD + nl] = acc[t][i];
        }
      }
    }
  }
}

// =====================================================================
// Bilinear downscale rate 2 (align_corners=False) == 2x2 average pool
// =====================================================================
__global__ __launch_bounds__(256)
void resize_pool_kernel(const float* __restrict__ in, float* __restrict__ out) {
  const int idx = blockIdx.x * 256 + threadIdx.x;      // 4*32*64*64 threads
  const int j = idx & 63, i = (idx >> 6) & 63, c = idx >> 12;
  const float* p = in + c * (128 * 128);
  const int r = 2 * i, s = 2 * j;
  out[idx] = 0.25f * (p[r * 128 + s] + p[r * 128 + s + 1] +
                      p[(r + 1) * 128 + s] + p[(r + 1) * 128 + s + 1]);
}

// =====================================================================
// GroupNorm(32 groups) + SiLU.  One block per (batch, group): 8ch x 4096px.
// =====================================================================
__global__ __launch_bounds__(256)
void groupnorm_silu_kernel(const float* __restrict__ in, const float* __restrict__ gamma,
                           const float* __restrict__ beta, float* __restrict__ out) {
  const int bg = blockIdx.x;
  const int b = bg >> 5, g = bg & 31;
  const int base = (b * 256 + g * 8) * HWD;
  float s = 0.f, q = 0.f;
  for (int i = threadIdx.x; i < 8 * HWD; i += 256) { const float v = in[base + i]; s += v; q += v * v; }
  __shared__ float rs[256], rq[256];
  rs[threadIdx.x] = s; rq[threadIdx.x] = q;
  __syncthreads();
  for (int st = 128; st > 0; st >>= 1) {
    if (threadIdx.x < st) { rs[threadIdx.x] += rs[threadIdx.x + st]; rq[threadIdx.x] += rq[threadIdx.x + st]; }
    __syncthreads();
  }
  const float mean = rs[0] * (1.f / 32768.f);
  const float var  = rq[0] * (1.f / 32768.f) - mean * mean;
  const float rstd = rsqrtf(var + 1e-6f);
  for (int i = threadIdx.x; i < 8 * HWD; i += 256) {
    const int cl = i >> 12;
    float v = (in[base + i] - mean) * rstd * gamma[g * 8 + cl] + beta[g * 8 + cl];
    out[base + i] = v / (1.f + __expf(-v));      // SiLU
  }
}

// =====================================================================
// Depthwise 7x7 conv, pad 3.  Block: 64x16 slab of one (b,c) plane.
// =====================================================================
__global__ __launch_bounds__(256)
void dwconv7_kernel(const float* __restrict__ in, const float* __restrict__ wgt,
                    const float* __restrict__ bias, float* __restrict__ out) {
  const int plane = blockIdx.x;           // b*256 + c
  const int th = blockIdx.y;              // rows [16*th, 16*th+16)
  const int c = plane & 255;
  const float* ip = in + plane * HWD;
  __shared__ float tile[22 * 70];
  __shared__ float wsm[49];
  if (threadIdx.x < 49) wsm[threadIdx.x] = wgt[c * 49 + threadIdx.x];
  for (int i = threadIdx.x; i < 22 * 70; i += 256) {
    const int r = i / 70, cc = i % 70;
    const int h = th * 16 + r - 3, w = cc - 3;
    tile[i] = ((unsigned)h < 64u && (unsigned)w < 64u) ? ip[h * 64 + w] : 0.f;
  }
  __syncthreads();
  const float bv = bias[c];
  for (int t = threadIdx.x; t < 1024; t += 256) {
    const int oh = t >> 6, ow = t & 63;
    float acc = bv;
#pragma unroll
    for (int i = 0; i < 7; ++i)
#pragma unroll
      for (int j = 0; j < 7; ++j)
        acc += tile[(oh + i) * 70 + (ow + j)] * wsm[i * 7 + j];
    out[plane * HWD + (th * 16 + oh) * 64 + ow] = acc;
  }
}

// =====================================================================
// Deformable sampling for tap k:
//   s[b, g*32+c, h, w] = sigmoid(mask) * bilinear(x[b,g*32+c], h+kh+dy, w+kw+dx)
// co [B,216,64,64]: offsets ch g*18+k*2(+1), masks ch 144+g*9+k.
// =====================================================================
__global__ __launch_bounds__(256)
void deform_sample_kernel(const float* __restrict__ x, const float* __restrict__ co,
                          float* __restrict__ s, int k) {
  const int idx = blockIdx.x * 256 + threadIdx.x;     // 4*8*4096 threads
  const int hw = idx & 4095, g = (idx >> 12) & 7, b = idx >> 15;
  const int h = hw >> 6, w = hw & 63;
  const int cbase = b * 216 * HWD + hw;
  const float dy = co[cbase + (g * 18 + k * 2)     * HWD];
  const float dx = co[cbase + (g * 18 + k * 2 + 1) * HWD];
  const float mk = co[cbase + (144 + g * 9 + k)    * HWD];
  const float m  = 1.f / (1.f + __expf(-mk));
  const float fy = (float)h + (float)(k / 3 - 1) + dy;
  const float fx = (float)w + (float)(k % 3 - 1) + dx;
  const float y0f = floorf(fy), x0f = floorf(fx);
  const int iy = (int)y0f, ix = (int)x0f;
  const float wy = fy - y0f, wx = fx - x0f;
  // zero-outside weights (avoid per-corner predicated loads: clamp index, zero weight)
  const bool vy0 = (unsigned)iy       < 64u, vy1 = (unsigned)(iy + 1) < 64u;
  const bool vx0 = (unsigned)ix       < 64u, vx1 = (unsigned)(ix + 1) < 64u;
  const float w00 = (vy0 && vx0) ? (1.f - wy) * (1.f - wx) : 0.f;
  const float w01 = (vy0 && vx1) ? (1.f - wy) * wx         : 0.f;
  const float w10 = (vy1 && vx0) ? wy * (1.f - wx)         : 0.f;
  const float w11 = (vy1 && vx1) ? wy * wx                 : 0.f;
  const int cy0 = min(max(iy, 0), 63),     cy1 = min(max(iy + 1, 0), 63);
  const int cx0 = min(max(ix, 0), 63),     cx1 = min(max(ix + 1, 0), 63);
  const int i00 = cy0 * 64 + cx0, i01 = cy0 * 64 + cx1;
  const int i10 = cy1 * 64 + cx0, i11 = cy1 * 64 + cx1;
  const int pbase = (b * 256 + g * 32) * HWD;
  for (int c = 0; c < 32; ++c) {
    const float* pl = x + pbase + c * HWD;
    const float v = w00 * pl[i00] + w01 * pl[i01] + w10 * pl[i10] + w11 * pl[i11];
    s[pbase + c * HWD + hw] = v * m;
  }
}

// =====================================================================
extern "C" void kernel_launch(void* const* d_in, const int* in_sizes, int n_in,
                              void* d_out, int out_size, void* d_ws, size_t ws_size,
                              hipStream_t stream) {
  const float* x_main  = (const float*)d_in[0];
  const float* inpfeat = (const float*)d_in[1];
  const float* w1a = (const float*)d_in[2];  const float* b1a = (const float*)d_in[3];
  const float* g1  = (const float*)d_in[4];  const float* be1 = (const float*)d_in[5];
  const float* w1b = (const float*)d_in[6];  const float* b1b = (const float*)d_in[7];
  const float* g2  = (const float*)d_in[8];  const float* be2 = (const float*)d_in[9];
  const float* w1c = (const float*)d_in[10]; const float* b1c = (const float*)d_in[11];
  const float* w2  = (const float*)d_in[12]; const float* b2  = (const float*)d_in[13];
  const float* g3  = (const float*)d_in[14]; const float* be3 = (const float*)d_in[15];
  const float* w_off = (const float*)d_in[16]; const float* b_off = (const float*)d_in[17];
  const float* w_dcn = (const float*)d_in[18]; const float* b_dcn = (const float*)d_in[19];

  float* out_warp = (float*)d_out;                      // [4,256,64,64]
  float* out_off  = out_warp + (size_t)4 * 256 * HWD;   // off_feat [4,256,64,64]

  float* ws   = (float*)d_ws;
  float* t_rs = ws;                         //   524288 : resized inpfeat [4,32,64,64]
  float* fA   = t_rs + 524288;              //  4194304
  float* fB   = fA + 4194304;               //  4194304
  float* coB  = fB + 4194304;               //  3538944 : [4,216,64,64]
  float* sbuf = coB + 3538944;              //  4194304 : deform samples

  const dim3 blk(256);
  const dim3 gN(NTOT / 128, 4);             // 128 n-blocks x 4 m-blocks (M=256 or 216)

  // 1) resize (avg-pool 2x2)
  resize_pool_kernel<<<524288 / 256, blk, 0, stream>>>(inpfeat, t_rs);
  // 2) 1x1 conv over concat([inp(32), x_main(256)]) -> fA
  gemm_tap_kernel<1, true><<<gN, blk, 0, stream>>>(w1a, 0, 256, 288, 32,
                                                   t_rs, x_main, 0, 0, b1a, nullptr, fA);
  // 3) GN + SiLU -> fB
  groupnorm_silu_kernel<<<128, blk, 0, stream>>>(fA, g1, be1, fB);
  // 4) depthwise 7x7 -> fA ; GN+SiLU -> fB
  dwconv7_kernel<<<dim3(1024, 4), blk, 0, stream>>>(fB, w1b, b1b, fA);
  groupnorm_silu_kernel<<<128, blk, 0, stream>>>(fA, g2, be2, fB);
  // 5) 1x1 conv -> fA
  gemm_tap_kernel<1, false><<<gN, blk, 0, stream>>>(w1c, 0, 256, 256, 256,
                                                    fB, nullptr, 0, 0, b1c, nullptr, fA);
  // 6) 3x3 conv (9 taps, accumulate in fB)
  for (int t = 0; t < 9; ++t)
    gemm_tap_kernel<9, false><<<gN, blk, 0, stream>>>(w2, t, 256, 256, 256,
                                                      fA, nullptr, t / 3 - 1, t % 3 - 1,
                                                      b2, t == 0 ? nullptr : fB, fB);
  // 7) GN + SiLU -> off_feat (second half of d_out)
  groupnorm_silu_kernel<<<128, blk, 0, stream>>>(fB, g3, be3, out_off);
  // 8) 3x3 offset/mask conv 256 -> 216 (9 taps, accumulate in coB)
  for (int t = 0; t < 9; ++t)
    gemm_tap_kernel<9, false><<<gN, blk, 0, stream>>>(w_off, t, 216, 256, 256,
                                                      out_off, nullptr, t / 3 - 1, t % 3 - 1,
                                                      b_off, t == 0 ? nullptr : coB, coB);
  // 9) modulated deformable conv: per tap, gather samples then GEMM-accumulate
  for (int k = 0; k < 9; ++k) {
    deform_sample_kernel<<<(4 * 8 * HWD) / 256, blk, 0, stream>>>(x_main, coB, sbuf, k);
    gemm_tap_kernel<9, false><<<gN, blk, 0, stream>>>(w_dcn, k, 256, 256, 256,
                                                      sbuf, nullptr, 0, 0,
                                                      b_dcn, k == 0 ? nullptr : out_warp, out_warp);
  }
}
